// _SAGE_59313498358200
// MI455X (gfx1250) — compile-verified, matrix operand order checked by reading, not run on hardware
//
#include <hip/hip_runtime.h>
#include <hip/hip_bf16.h>

// ---------------------------------------------------------------------------
// 2-layer GraphSAGE (mean aggregator) for MI455X / gfx1250.
//  Phase 1: degree count + inverse (f32 atomics)
//  Phase 2: weight transpose + f32->bf16 convert (weights stay L2-resident)
//  Phase 3: edge scatter-sum (global_atomic_add_f32, wave-per-edge)
//  Phase 4: fused dual-GEMM  out = x@Ws + (neigh*invdeg)@Wn + b  (+ReLU)
//           via v_wmma_f32_16x16x32_bf16, two 16x16 tiles per wave, K=256.
// ---------------------------------------------------------------------------

typedef __bf16 bf16;
typedef bf16  bf16x8 __attribute__((ext_vector_type(8)));
typedef bf16  v16bf  __attribute__((ext_vector_type(16)));
typedef float v8f    __attribute__((ext_vector_type(8)));

__device__ __forceinline__ bf16 f32_to_bf16(float f) {
    union { float f; unsigned u; } v; v.f = f;
    unsigned r = v.u + 0x7FFFu + ((v.u >> 16) & 1u);   // round-to-nearest-even
    unsigned short h = (unsigned short)(r >> 16);
    bf16 b; __builtin_memcpy(&b, &h, sizeof(h));
    return b;
}

// -------------------------------- utility ----------------------------------

__global__ void _sage_zero_f32(float* __restrict__ p, long long n4) {
    long long i = (long long)blockIdx.x * blockDim.x + threadIdx.x;
    long long stride = (long long)gridDim.x * blockDim.x;
    float4 z = make_float4(0.f, 0.f, 0.f, 0.f);
    for (; i < n4; i += stride) ((float4*)p)[i] = z;
}

__global__ void _sage_copy_f32(const float* __restrict__ src, float* __restrict__ dst,
                               long long n4) {
    long long i = (long long)blockIdx.x * blockDim.x + threadIdx.x;
    long long stride = (long long)gridDim.x * blockDim.x;
    for (; i < n4; i += stride) ((float4*)dst)[i] = ((const float4*)src)[i];
}

// -------------------------------- phase 1 ----------------------------------

__global__ void _sage_degree(const int* __restrict__ dst, float* __restrict__ deg, int nE) {
    int i = blockIdx.x * blockDim.x + threadIdx.x;
    if (i < nE) atomicAdd(&deg[dst[i]], 1.0f);
}

__global__ void _sage_invdeg(float* __restrict__ deg, int n) {
    int i = blockIdx.x * blockDim.x + threadIdx.x;
    if (i < n) {
        float d = deg[i];
        deg[i] = (d > 0.f) ? (1.0f / d) : 0.0f;
    }
}

// ------------------------- phase 2: weight transpose -----------------------
// Input  W  : [K=256][N=256] f32 (row-major, fan_in x fan_out)
// Output Wt : [N=256][K=256] bf16  -> per-lane WMMA B fetches are contiguous.

__global__ void _sage_wt_bf16(const float* __restrict__ W, bf16* __restrict__ Wt) {
    int idx = blockIdx.x * blockDim.x + threadIdx.x;   // 0 .. 65535
    int n = idx >> 8;
    int k = idx & 255;
    Wt[(size_t)n * 256 + k] = f32_to_bf16(W[(size_t)k * 256 + n]);
}

// ------------------------- phase 3: edge scatter ---------------------------
// One wave32 per edge; lane handles 8 columns with stride-32 for coalescing.

__global__ void _sage_scatter(const float* __restrict__ x,
                              const int* __restrict__ src,
                              const int* __restrict__ dst,
                              float* __restrict__ neigh, int nE) {
    int e = blockIdx.x * 8 + (threadIdx.x >> 5);
    if (e >= nE) return;
    int lane = threadIdx.x & 31;
    const float* __restrict__ xr = x + (size_t)src[e] * 256;
    float* __restrict__ nr = neigh + (size_t)dst[e] * 256;
#pragma unroll
    for (int i = 0; i < 8; ++i) {
        int c = lane + i * 32;
        atomicAdd(nr + c, xr[c]);
    }
}

// --------------------- phase 4: fused dual-GEMM (WMMA) ---------------------
// Block = 256 threads = 8 waves, handles 16 rows x 256 cols.
// Stage self-activations and degree-normalized neighbor sums as bf16 in LDS,
// each wave computes two 16x16 tiles: out = A_s*B_s + A_n*B_n + bias (+relu).

__global__ __launch_bounds__(256)
void _sage_gemm(const float* __restrict__ x,
                const float* __restrict__ neigh,
                const float* __restrict__ inv_deg,
                const bf16* __restrict__ WtS,
                const bf16* __restrict__ WtN,
                const float* __restrict__ bias,
                float* __restrict__ out,
                int N, int do_relu) {
    __shared__ bf16 sA[2][16][256];                 // [self|neigh][row][k], 16 KB

    const int t    = threadIdx.x;
    const int row0 = blockIdx.x * 16;

    // ---- stage A tiles: 16 rows x 256 K, f32 -> bf16 (neigh scaled by 1/deg)
    {
        int row = t >> 4;                           // 0..15
        int cb  = (t & 15) << 4;                    // 0..240 step 16
        int gr  = row0 + row;
        if (gr >= N) gr = N - 1;                    // clamp (tail-safe)
        const float* xs = x     + (size_t)gr * 256 + cb;
        const float* xn = neigh + (size_t)gr * 256 + cb;
        float inv = inv_deg[gr];
#pragma unroll
        for (int i = 0; i < 16; i += 4) {
            float4 v = *(const float4*)(xs + i);
            sA[0][row][cb + i + 0] = f32_to_bf16(v.x);
            sA[0][row][cb + i + 1] = f32_to_bf16(v.y);
            sA[0][row][cb + i + 2] = f32_to_bf16(v.z);
            sA[0][row][cb + i + 3] = f32_to_bf16(v.w);
            float4 w = *(const float4*)(xn + i);
            sA[1][row][cb + i + 0] = f32_to_bf16(w.x * inv);
            sA[1][row][cb + i + 1] = f32_to_bf16(w.y * inv);
            sA[1][row][cb + i + 2] = f32_to_bf16(w.z * inv);
            sA[1][row][cb + i + 3] = f32_to_bf16(w.w * inv);
        }
    }
    __syncthreads();

    const int wave = t >> 5;
    const int lane = t & 31;
    const int half = lane >> 4;                     // 0: lanes 0-15, 1: lanes 16-31
    const int l16  = lane & 15;

#pragma unroll
    for (int ct = 0; ct < 2; ++ct) {
        const int col0 = (wave * 2 + ct) * 16;
        v8f acc = {};

#pragma unroll
        for (int m = 0; m < 2; ++m) {
            const bf16* __restrict__ Wt = m ? WtN : WtS;
            const bf16* __restrict__ brow = Wt + (size_t)(col0 + l16) * 256;
#pragma unroll
            for (int k8 = 0; k8 < 8; ++k8) {
                const int kb = k8 * 32 + half * 8;  // per-lane K base (ISA layout)

                // A: row = l16; elements 0..7 at kb, 8..15 at kb+16 (two b128 DS loads)
                bf16x8 a_lo = *(const bf16x8*)&sA[m][l16][kb];
                bf16x8 a_hi = *(const bf16x8*)&sA[m][l16][kb + 16];
                v16bf A = __builtin_shufflevector(a_lo, a_hi,
                            0,1,2,3,4,5,6,7,8,9,10,11,12,13,14,15);

                // B: col = l16 of this tile; same K layout, contiguous in Wt[n][k]
                bf16x8 b_lo = *(const bf16x8*)(brow + kb);
                bf16x8 b_hi = *(const bf16x8*)(brow + kb + 16);
                v16bf B = __builtin_shufflevector(b_lo, b_hi,
                            0,1,2,3,4,5,6,7,8,9,10,11,12,13,14,15);

                __builtin_prefetch(brow + kb + 32, 0, 3);  // next K slice -> L2/L0

                acc = __builtin_amdgcn_wmma_f32_16x16x32_bf16(
                          false, A, false, B, (short)0, acc, false, false);
            }
        }

        // C/D layout: VGPR r -> row = r + 8*half, col = l16
        const int   gc = col0 + l16;
        const float bv = bias[gc];
#pragma unroll
        for (int r = 0; r < 8; ++r) {
            int gr = row0 + r + half * 8;
            if (gr < N) {
                float v = acc[r] + bv;
                if (do_relu) v = fmaxf(v, 0.0f);
                out[(size_t)gr * 256 + gc] = v;
            }
        }
    }
}

// ------------------------------ host launch --------------------------------

extern "C" void kernel_launch(void* const* d_in, const int* in_sizes, int n_in,
                              void* d_out, int out_size, void* d_ws, size_t ws_size,
                              hipStream_t stream) {
    const float* feat = (const float*)d_in[0];
    const int*   src  = (const int*)d_in[1];
    const int*   dst  = (const int*)d_in[2];
    const float* Ws0  = (const float*)d_in[3];
    const float* Wn0  = (const float*)d_in[4];
    const float* b0   = (const float*)d_in[5];
    const float* Ws1  = (const float*)d_in[6];
    const float* Wn1  = (const float*)d_in[7];
    const float* b1   = (const float*)d_in[8];

    const int N  = in_sizes[0] / 256;       // 100000
    const int nE = in_sizes[1];             // 1600000

    float* out   = (float*)d_out;
    float* out_x0 = out;                              // x0 passthrough
    float* out_a1 = out + (size_t)N * 256;            // relu(conv1)
    float* out_h2 = out + (size_t)2 * N * 256;        // conv2

    // workspace layout (256B-aligned sections)
    char* ws = (char*)d_ws;
    size_t degBytes = (((size_t)N * 4 + 255) / 256) * 256;
    float* deg  = (float*)ws;                         // becomes inv_deg in place
    bf16*  WtS0 = (bf16*)(ws + degBytes);
    bf16*  WtN0 = WtS0 + 256 * 256;
    bf16*  WtS1 = WtN0 + 256 * 256;
    bf16*  WtN1 = WtS1 + 256 * 256;
    float* neigh = (float*)(ws + degBytes + 4 * 256 * 256 * sizeof(bf16));

    const long long featElems = (long long)N * 256;

    // -- degree + inverse degree
    _sage_zero_f32<<<1024, 256, 0, stream>>>(deg, (long long)N / 4);
    _sage_degree<<<(nE + 255) / 256, 256, 0, stream>>>(dst, deg, nE);
    _sage_invdeg<<<(N + 255) / 256, 256, 0, stream>>>(deg, N);

    // -- bf16 transposed weights (L2-resident for all GEMM blocks)
    _sage_wt_bf16<<<256, 256, 0, stream>>>(Ws0, WtS0);
    _sage_wt_bf16<<<256, 256, 0, stream>>>(Wn0, WtN0);
    _sage_wt_bf16<<<256, 256, 0, stream>>>(Ws1, WtS1);
    _sage_wt_bf16<<<256, 256, 0, stream>>>(Wn1, WtN1);

    // -- output[0] = feat (passthrough)
    _sage_copy_f32<<<2048, 256, 0, stream>>>(feat, out_x0, featElems / 4);

    const int gemmBlocks = (N + 15) / 16;
    const int edgeBlocks = (nE + 7) / 8;

    // ---- layer 0: neigh = scatter_sum(feat); a1 = relu(feat@Ws0 + mean@Wn0 + b0)
    _sage_zero_f32<<<4096, 256, 0, stream>>>(neigh, featElems / 4);
    _sage_scatter<<<edgeBlocks, 256, 0, stream>>>(feat, src, dst, neigh, nE);
    _sage_gemm<<<gemmBlocks, 256, 0, stream>>>(feat, neigh, deg, WtS0, WtN0, b0,
                                               out_a1, N, /*relu=*/1);

    // ---- layer 1: neigh = scatter_sum(a1); h2 = a1@Ws1 + mean@Wn1 + b1
    _sage_zero_f32<<<4096, 256, 0, stream>>>(neigh, featElems / 4);
    _sage_scatter<<<edgeBlocks, 256, 0, stream>>>(out_a1, src, dst, neigh, nE);
    _sage_gemm<<<gemmBlocks, 256, 0, stream>>>(out_a1, neigh, deg, WtS1, WtN1, b1,
                                               out_h2, N, /*relu=*/0);
}
